// KinomeGNN_10720238371004
// MI455X (gfx1250) — matrix operand order repaired and checked by reading, not compiled, not run on hardware
//
#include <hip/hip_runtime.h>
#include <hip/hip_bf16.h>

// ---------------------------------------------------------------------------
// KinomeGNN forward on MI455X (gfx1250, wave32).
// HID == 32 == wave32: one lane per feature everywhere.
// Dense [N,32]x[32,32] GEMMs use v_wmma_f32_16x16x32_f16 (f32 accumulate),
// fused with BatchNorm statistic accumulation. BN+ReLU are applied on the
// fly by consumers (no extra N*32 passes). All random-index arrays are
// L2-resident (192MB L2); HBM traffic is dominated by the edge-list streams.
// ---------------------------------------------------------------------------

typedef __attribute__((ext_vector_type(16))) _Float16 v16h;
typedef __attribute__((ext_vector_type(8)))  float    v8f;

#define NN   200000      // nodes
#define NE   6400000     // edges
#define NG   2048        // graphs
#define HID  32
#define EPSF 1e-5f

// ---------------------------------------------------------------- utilities
__device__ __forceinline__ unsigned enc_f32(float f) {
    unsigned u = __float_as_uint(f);
    return (u & 0x80000000u) ? ~u : (u | 0x80000000u);
}
__device__ __forceinline__ float dec_f32(unsigned e) {
    return (e & 0x80000000u) ? __uint_as_float(e ^ 0x80000000u)
                             : __uint_as_float(~e);
}
__device__ __forceinline__ unsigned pack_h2(float a, float b) {
    _Float16 ha = (_Float16)a, hb = (_Float16)b;
    unsigned short ua, ub;
    __builtin_memcpy(&ua, &ha, 2);
    __builtin_memcpy(&ub, &hb, 2);
    return ((unsigned)ub << 16) | (unsigned)ua;
}

// ------------------------------------------------------------------- kernels
__global__ void k_zero(float* p, long n) {
    long i = (long)blockIdx.x * blockDim.x + threadIdx.x;
    long stride = (long)gridDim.x * blockDim.x;
    for (; i < n; i += stride) p[i] = 0.0f;
}

// Pack W2l / W2r (stored [32 out][32 in] row-major) into the WMMA B-operand
// layout: lane l holds column n = l&15; lane-half h holds K = 16h..16h+15,
// two f16 per dword (element 2v = low half of dword v).
__global__ void k_pack_w(const float* __restrict__ W2l,
                         const float* __restrict__ W2r,
                         unsigned* __restrict__ pl,
                         unsigned* __restrict__ pr) {
    int t = threadIdx.x;            // 64 threads: [tile(2)][lane(32)]
    if (t >= 64) return;
    int tile = t >> 5, lane = t & 31;
    int n = tile * 16 + (lane & 15);
    int half = (lane >> 4) & 1;
    #pragma unroll
    for (int v = 0; v < 8; ++v) {
        int k = 16 * half + 2 * v;
        pl[(size_t)t * 8 + v] = pack_h2(W2l[n * 32 + k], W2l[n * 32 + k + 1]);
        pr[(size_t)t * 8 + v] = pack_h2(W2r[n * 32 + k], W2r[n * 32 + k + 1]);
    }
}

// Layer-1 aggregation: scalar feature. One thread per edge.
__global__ void k_scatter1(const float* __restrict__ x,
                           const int* __restrict__ ei,
                           float* __restrict__ agg1,
                           float* __restrict__ deg) {
    int e = blockIdx.x * blockDim.x + threadIdx.x;
    if (e >= NE) return;
    int s = ei[e], d = ei[NE + e];
    atomicAdd(&agg1[d], x[s]);
    atomicAdd(&deg[d], 1.0f);
}

// h1pre[n][c] = (agg1[n]/max(deg,1))*W1l[c] + b1l[c] + x[n]*W1r[c]
// + block-local BN1 statistics (LDS f32 atomics) -> global sums.
__global__ void k_node1(const float* __restrict__ x,
                        const float* __restrict__ agg1,
                        const float* __restrict__ deg,
                        const float* __restrict__ W1l,
                        const float* __restrict__ b1l,
                        const float* __restrict__ W1r,
                        float* __restrict__ h1pre,
                        float* __restrict__ sums1) {   // [0..31]=sum [32..63]=sumsq
    __shared__ float ssum[32], ssq[32];
    int t = threadIdx.x;                 // 256 = 8 nodes x 32 feats
    if (t < 32) { ssum[t] = 0.f; ssq[t] = 0.f; }
    __syncthreads();
    int c = t & 31;
    int n = blockIdx.x * 8 + (t >> 5);
    if (n < NN) {
        float inv = 1.0f / fmaxf(deg[n], 1.0f);
        float a = agg1[n] * inv;
        float v = a * W1l[c] + b1l[c] + x[n] * W1r[c];
        h1pre[(size_t)n * 32 + c] = v;
        atomicAdd(&ssum[c], v);
        atomicAdd(&ssq[c], v * v);
    }
    __syncthreads();
    if (t < 32) {
        atomicAdd(&sums1[t], ssum[t]);
        atomicAdd(&sums1[32 + t], ssq[t]);
    }
}

// scale = gamma * rsqrt(var + eps);  shift = beta - mean*scale
__global__ void k_bn_params(const float* __restrict__ sums,
                            const float* __restrict__ gamma,
                            const float* __restrict__ beta,
                            float* __restrict__ scale,
                            float* __restrict__ shift) {
    int c = threadIdx.x;
    if (c >= 32) return;
    const float invN = 1.0f / (float)NN;
    float mu  = sums[c] * invN;
    float var = sums[32 + c] * invN - mu * mu;
    float sc  = gamma[c] * rsqrtf(var + EPSF);
    scale[c] = sc;
    shift[c] = beta[c] - mu * sc;
}

// Layer-2 aggregation: one wave per edge, one lane per feature.
// BN1+ReLU applied on the fly (h1 never materialized post-activation).
__global__ void k_scatter2(const int* __restrict__ ei,
                           const float* __restrict__ h1pre,
                           const float* __restrict__ s1,
                           const float* __restrict__ sh1,
                           float* __restrict__ agg2) {
    int tid = blockIdx.x * blockDim.x + threadIdx.x;
    int e = tid >> 5, c = tid & 31;
    if (e >= NE) return;
    int s = ei[e], d = ei[NE + e];
    float v = fmaxf(h1pre[(size_t)s * 32 + c] * s1[c] + sh1[c], 0.0f);
    atomicAdd(&agg2[(size_t)d * 32 + c], v);
}

// Layer-2 dense transform via WMMA. One wave per 16-node tile:
//   h2pre[16,32] = (agg2/deg)[16,32] @ W2l^T + relu(bn1(h1pre))[16,32] @ W2r^T + b2l
// 4 x v_wmma_f32_16x16x32_f16 per tile (accumulator chained, seeded w/ bias).
// BN2 sum/sumsq accumulated from the f32 accumulator in-register.
__global__ void k_gemm2(const float* __restrict__ agg2,
                        const float* __restrict__ deg,
                        const float* __restrict__ h1pre,
                        const float* __restrict__ s1,
                        const float* __restrict__ sh1,
                        const float* __restrict__ b2l,
                        const unsigned* __restrict__ pl,
                        const unsigned* __restrict__ pr,
                        float* __restrict__ h2pre,
                        float* __restrict__ sums2) {
    int lane = threadIdx.x & 31;
    int wave = threadIdx.x >> 5;
    int tile = blockIdx.x * 8 + wave;
    if (tile >= NN / 16) return;                 // whole-wave uniform exit

    int m    = lane & 15;                        // A row held by this lane
    int half = lane >> 4;
    size_t node = (size_t)tile * 16 + m;
    float inv = 1.0f / fmaxf(deg[node], 1.0f);

    // A operand per-lane layout: elems 0..7 = K in [16h, 16h+8),
    //                            elems 8..15 = K in [16h+8, 16h+16)
    int kb0 = 16 * half, kb1 = 16 * half + 8;
    v16h A1, A2;
    #pragma unroll
    for (int i = 0; i < 8; ++i) {
        int k0 = kb0 + i, k1 = kb1 + i;
        A1[i]     = (_Float16)(agg2[node * 32 + k0] * inv);
        A1[8 + i] = (_Float16)(agg2[node * 32 + k1] * inv);
        A2[i]     = (_Float16)fmaxf(h1pre[node * 32 + k0] * s1[k0] + sh1[k0], 0.f);
        A2[8 + i] = (_Float16)fmaxf(h1pre[node * 32 + k1] * s1[k1] + sh1[k1], 0.f);
    }

    #pragma unroll
    for (int t = 0; t < 2; ++t) {                // two 16-wide column tiles
        int ncol = t * 16 + (lane & 15);
        v16h Bl = *(const v16h*)(pl + ((size_t)t * 32 + lane) * 8);
        v16h Br = *(const v16h*)(pr + ((size_t)t * 32 + lane) * 8);

        v8f acc;
        float bias = b2l[ncol];
        #pragma unroll
        for (int r = 0; r < 8; ++r) acc[r] = bias;

        acc = __builtin_amdgcn_wmma_f32_16x16x32_f16(
                  false, A1, false, Bl, (short)0, acc, false, false);
        acc = __builtin_amdgcn_wmma_f32_16x16x32_f16(
                  false, A2, false, Br, (short)0, acc, false, false);

        float sum = 0.f, sq = 0.f;
        #pragma unroll
        for (int r = 0; r < 8; ++r) {            // D: lane=col, row = r + 8*half
            float d = acc[r];
            size_t row = (size_t)tile * 16 + r + 8 * half;
            h2pre[row * 32 + ncol] = d;
            sum += d; sq += d * d;
        }
        atomicAdd(&sums2[ncol], sum);
        atomicAdd(&sums2[32 + ncol], sq);
    }
}

// Gate score per node (wave per node, shuffle reduce) + segment max
// via order-preserving uint atomicMax.
__global__ void k_score(const float* __restrict__ h2pre,
                        const float* __restrict__ s2,
                        const float* __restrict__ sh2,
                        const float* __restrict__ gate_w,
                        const float* __restrict__ gate_b,
                        const int* __restrict__ batch,
                        float* __restrict__ score,
                        unsigned* __restrict__ smax) {
    int lane = threadIdx.x & 31;
    int n = blockIdx.x * 8 + (threadIdx.x >> 5);
    if (n >= NN) return;
    float v = fmaxf(h2pre[(size_t)n * 32 + lane] * s2[lane] + sh2[lane], 0.f);
    float p = v * gate_w[lane];
    #pragma unroll
    for (int o = 16; o; o >>= 1) p += __shfl_xor(p, o, 32);
    if (lane == 0) {
        float sc = p + gate_b[0];
        score[n] = sc;
        atomicMax(&smax[batch[n]], enc_f32(sc));
    }
}

__global__ void k_exp(const float* __restrict__ score,
                      const unsigned* __restrict__ smax,
                      const int* __restrict__ batch,
                      float* __restrict__ exsc,
                      float* __restrict__ denom) {
    int n = blockIdx.x * blockDim.x + threadIdx.x;
    if (n >= NN) return;
    int b = batch[n];
    float ex = __expf(score[n] - dec_f32(smax[b]));
    exsc[n] = ex;
    atomicAdd(&denom[b], ex);
}

// g[b][c] += (ex/denom) * relu(bn2(h2pre))  — wave per node, lane per feature.
__global__ void k_pool(const float* __restrict__ h2pre,
                       const float* __restrict__ s2,
                       const float* __restrict__ sh2,
                       const float* __restrict__ exsc,
                       const float* __restrict__ denom,
                       const int* __restrict__ batch,
                       float* __restrict__ gacc) {
    int c = threadIdx.x & 31;
    int n = blockIdx.x * 8 + (threadIdx.x >> 5);
    if (n >= NN) return;
    int b = batch[n];
    float alpha = exsc[n] / denom[b];
    float v = fmaxf(h2pre[(size_t)n * 32 + c] * s2[c] + sh2[c], 0.f);
    atomicAdd(&gacc[(size_t)b * 32 + c], alpha * v);
}

__global__ void k_out(const float* __restrict__ gacc,
                      const float* __restrict__ lin_w,
                      const float* __restrict__ lin_b,
                      float* __restrict__ out) {
    int lane = threadIdx.x & 31;
    int g = blockIdx.x * 8 + (threadIdx.x >> 5);
    if (g >= NG) return;
    float p = gacc[(size_t)g * 32 + lane] * lin_w[lane];
    #pragma unroll
    for (int o = 16; o; o >>= 1) p += __shfl_xor(p, o, 32);
    if (lane == 0) {
        float z = p + lin_b[0];
        out[g] = 1.0f / (1.0f + __expf(-z));
    }
}

// ---------------------------------------------------------------------------
extern "C" void kernel_launch(void* const* d_in, const int* in_sizes, int n_in,
                              void* d_out, int out_size, void* d_ws, size_t ws_size,
                              hipStream_t stream) {
    const float* x      = (const float*)d_in[0];
    const int*   ei     = (const int*)  d_in[1];   // [2*E]
    const int*   batch  = (const int*)  d_in[2];
    const float* W1l    = (const float*)d_in[3];
    const float* b1l    = (const float*)d_in[4];
    const float* W1r    = (const float*)d_in[5];
    const float* W2l    = (const float*)d_in[6];
    const float* b2l    = (const float*)d_in[7];
    const float* W2r    = (const float*)d_in[8];
    const float* g1     = (const float*)d_in[9];
    const float* be1    = (const float*)d_in[10];
    const float* g2     = (const float*)d_in[11];
    const float* be2    = (const float*)d_in[12];
    const float* gate_w = (const float*)d_in[13];
    const float* gate_b = (const float*)d_in[14];
    const float* lin_w  = (const float*)d_in[15];
    const float* lin_b  = (const float*)d_in[16];
    float* out = (float*)d_out;

    // Workspace layout (floats). Zero-initialized region first (~27.5 MB).
    float* W = (float*)d_ws;
    size_t o = 0;
    float*    agg1  = W + o; o += NN;
    float*    deg   = W + o; o += NN;
    float*    agg2  = W + o; o += (size_t)NN * 32;
    float*    gacc  = W + o; o += (size_t)NG * 32;
    float*    denom = W + o; o += NG;
    unsigned* smax  = (unsigned*)(W + o); o += NG;
    float*    sums1 = W + o; o += 64;
    float*    sums2 = W + o; o += 64;
    const long ZCOUNT = (long)o;              // everything above gets zeroed
    float*    h1pre = W + o; o += (size_t)NN * 32;
    float*    h2pre = W + o; o += (size_t)NN * 32;
    float*    score = W + o; o += NN;
    float*    exsc  = W + o; o += NN;
    float*    s1    = W + o; o += 32;
    float*    sh1   = W + o; o += 32;
    float*    s2    = W + o; o += 32;
    float*    sh2   = W + o; o += 32;
    unsigned* pl    = (unsigned*)(W + o); o += 512;
    unsigned* pr    = (unsigned*)(W + o); o += 512;
    (void)ws_size; (void)n_in; (void)in_sizes; (void)out_size;

    k_zero<<<4096, 256, 0, stream>>>(W, ZCOUNT);
    k_pack_w<<<1, 64, 0, stream>>>(W2l, W2r, pl, pr);

    // Layer 1
    k_scatter1<<<NE / 256, 256, 0, stream>>>(x, ei, agg1, deg);
    k_node1<<<NN / 8, 256, 0, stream>>>(x, agg1, deg, W1l, b1l, W1r, h1pre, sums1);
    k_bn_params<<<1, 32, 0, stream>>>(sums1, g1, be1, s1, sh1);

    // Layer 2 (BN1+ReLU fused into the gather)
    k_scatter2<<<(NE * 32) / 256, 256, 0, stream>>>(ei, h1pre, s1, sh1, agg2);
    k_gemm2<<<(NN / 16 + 7) / 8, 256, 0, stream>>>(agg2, deg, h1pre, s1, sh1,
                                                   b2l, pl, pr, h2pre, sums2);
    k_bn_params<<<1, 32, 0, stream>>>(sums2, g2, be2, s2, sh2);

    // Attentional pooling + readout
    k_score<<<NN / 8, 256, 0, stream>>>(h2pre, s2, sh2, gate_w, gate_b,
                                        batch, score, smax);
    k_exp<<<(NN + 255) / 256, 256, 0, stream>>>(score, smax, batch, exsc, denom);
    k_pool<<<NN / 8, 256, 0, stream>>>(h2pre, s2, sh2, exsc, denom, batch, gacc);
    k_out<<<NG / 8, 256, 0, stream>>>(gacc, lin_w, lin_b, out);
}
